// GraphormerMultiHeadAttention_73839077752852
// MI455X (gfx1250) — matrix operand order repaired and checked by reading, not compiled, not run on hardware
//
#include <hip/hip_runtime.h>

// GraphormerMultiHeadAttention for MI455X (gfx1250, wave32, WMMA bf16)
// N=2048 nodes, D=512 hidden, H=8 heads.
// Workspace usage: ~98.5 MB.

typedef __bf16 bf16;
typedef __attribute__((ext_vector_type(16))) __bf16 v16bf;
typedef __attribute__((ext_vector_type(8)))  __bf16 v8bf;
typedef __attribute__((ext_vector_type(8)))  float  v8f;

#define NN 2048
#define DD 512
#define HH 8

__device__ __forceinline__ unsigned short f2bf_bits(float f) {
  union { float f; unsigned u; } x; x.f = f;
  unsigned r = x.u + 0x7FFFu + ((x.u >> 16) & 1u);   // round-to-nearest-even
  return (unsigned short)(r >> 16);
}

// ---- CDNA5 async global->LDS copy (16B per lane), tracked by ASYNCcnt ----
__device__ __forceinline__ void async_copy_b128(void* lds_dst, const void* gsrc) {
  unsigned lds_addr = (unsigned)(size_t)lds_dst;   // low 32 bits = LDS offset
  asm volatile("global_load_async_to_lds_b128 %0, %1, off"
               :: "v"(lds_addr), "v"(gsrc) : "memory");
}
__device__ __forceinline__ void wait_async0() {
#if __has_builtin(__builtin_amdgcn_s_wait_asynccnt)
  __builtin_amdgcn_s_wait_asynccnt(0);
#else
  asm volatile("s_wait_asynccnt 0x0" ::: "memory");
#endif
}

// A-fragment 16x32 bf16 (row-major source, stride in elements).
// lanes 0-15: row=lane, K in {0..7, 16..23}; lanes 16-31: row=lane-16, K in {8..15, 24..31}
__device__ __forceinline__ v16bf load_a_frag(const bf16* base, int stride, int lane) {
  int r  = lane & 15;
  int hi = (lane >> 4) & 1;
  const bf16* p = base + r * stride + hi * 8;
  union { v16bf v; v8bf h[2]; } u;
  u.h[0] = *(const v8bf*)(p);
  u.h[1] = *(const v8bf*)(p + 16);
  return u.v;
}

// B-fragment 32x16 bf16 from an N-major (B-transposed) source:
// element (k,n) at base[n*stride + k]. lane holds column n=lane&15,
// K in {0..7, 16..23} (lanes 0-15) or {8..15, 24..31} (lanes 16-31), matching A.
__device__ __forceinline__ v16bf load_b_frag(const bf16* base, int stride, int lane) {
  int n  = lane & 15;
  int hi = (lane >> 4) & 1;
  const bf16* p = base + n * stride + hi * 8;
  union { v16bf v; v8bf h[2]; } u;
  u.h[0] = *(const v8bf*)(p);
  u.h[1] = *(const v8bf*)(p + 16);
  return u.v;
}

#define WMMA_BF16(a, b, c) \
  __builtin_amdgcn_wmma_f32_16x16x32_bf16(false, (a), false, (b), (short)0, (c), false, false)

// ---------------------------------------------------------------- prep kernels
__global__ void cvt_kernel(const float* __restrict__ s, bf16* __restrict__ d, int n) {
  int i = blockIdx.x * blockDim.x + threadIdx.x;
  int st = gridDim.x * blockDim.x;
  unsigned short* ds = (unsigned short*)d;
  for (; i < n; i += st) ds[i] = f2bf_bits(s[i]);
}

// W is KxN row-major fp32; Wt is NxK row-major bf16 (Wt[n][k] = W[k][n])
__global__ void cvt_tr_kernel(const float* __restrict__ W, bf16* __restrict__ Wt, int K, int N) {
  int i = blockIdx.x * blockDim.x + threadIdx.x;
  int st = gridDim.x * blockDim.x;
  int total = K * N;
  unsigned short* ds = (unsigned short*)Wt;
  for (; i < total; i += st) {
    int n = i / K;
    int k = i - n * K;
    ds[i] = f2bf_bits(W[(size_t)k * N + n]);
  }
}

__global__ void bias_kernel(const float* __restrict__ sp, const float* __restrict__ ed,
                            float* __restrict__ b, int n) {
  int i = blockIdx.x * blockDim.x + threadIdx.x;
  int st = gridDim.x * blockDim.x;
  for (; i < n; i += st) {
    float s = sp[i];
    b[i] = (s == 0.0f) ? -1.0e6f : (s + ed[i]);
  }
}

// ---------------------------------------------------------------- GEMM (bf16 WMMA)
// C(MxN) = alpha * A(MxK) @ B(KxN);  A bf16 row-major, Bt = B^T bf16 (NxK row-major).
// MODE 0: bf16 row-major out; MODE 1: f32 row-major out;
// MODE 2: bf16 out scattered into per-head-transposed V layout:
//         flat f = r*4096+c -> h=f>>20, n=(f>>9)&2047, d=f&511; idx = (h<<20)+(d<<11)+n
template <int MODE>
__global__ void __launch_bounds__(256)
gemm_bf16_kernel(const bf16* __restrict__ A, const bf16* __restrict__ Bt,
                 void* __restrict__ Cout, int M, int N, int K, float alpha) {
  const int lane = threadIdx.x & 31;
  const int w    = threadIdx.x >> 5;               // 8 waves: 2 (M) x 4 (N)
  const int row0 = blockIdx.y * 64  + (w & 1) * 32;
  const int col0 = blockIdx.x * 256 + (w >> 1) * 64;

  v8f acc[2][4] = {};

  for (int k0 = 0; k0 < K; k0 += 32) {
    v16bf a0 = load_a_frag(A + (size_t)(row0)      * K + k0, K, lane);
    v16bf a1 = load_a_frag(A + (size_t)(row0 + 16) * K + k0, K, lane);
    v16bf b[4];
#pragma unroll
    for (int j = 0; j < 4; ++j)
      b[j] = load_b_frag(Bt + (size_t)(col0 + 16 * j) * K + k0, K, lane);
#pragma unroll
    for (int j = 0; j < 4; ++j) {
      acc[0][j] = WMMA_BF16(a0, b[j], acc[0][j]);
      acc[1][j] = WMMA_BF16(a1, b[j], acc[1][j]);
    }
  }

  const int cn = lane & 15;
  const int hi = (lane >> 4) & 1;
#pragma unroll
  for (int i = 0; i < 2; ++i)
#pragma unroll
    for (int j = 0; j < 4; ++j)
#pragma unroll
      for (int e = 0; e < 8; ++e) {
        int r = row0 + 16 * i + e + 8 * hi;
        int c = col0 + 16 * j + cn;
        float v = acc[i][j][e] * alpha;
        if (MODE == 0) {
          ((unsigned short*)Cout)[(size_t)r * N + c] = f2bf_bits(v);
        } else if (MODE == 1) {
          ((float*)Cout)[(size_t)r * N + c] = v;
        } else {
          size_t idx = ((size_t)(r >> 8) << 20) + ((size_t)(c & 511) << 11)
                     + ((size_t)(r & 255) << 3) + (size_t)(c >> 9);
          ((unsigned short*)Cout)[idx] = f2bf_bits(v);
        }
      }
}

// ---------------------------------------------------------------- attention
// Per (head, 64-row block). Flash-style online softmax over key chunks of 64.
// Q async-staged in LDS (pre-scaled by 1/sqrt(D)); K/Vt bf16 read from L2 with
// a dup=1 wave->fragment assignment; bias f32 from L2 (prefetched).
// Stage 1: wave w -> m-tile (w&3), k-half (w>>2), all 4 row-tiles; the two
//          k-halves land in Ssa/Ssb and are summed (with bias) in softmax.
// Stage 2: wave w -> all 4 row-tiles x unique 64-wide d-slice (dup=1 on Vt).
// Output AO[n][h*512+d] bf16 (= transpose(1,0,2).reshape of reference).
__global__ void __launch_bounds__(256)
attn_kernel(const bf16* __restrict__ Q, const bf16* __restrict__ Km,
            const bf16* __restrict__ Vt, const float* __restrict__ biasM,
            bf16* __restrict__ AO) {
  __shared__ __align__(16) bf16  Qs[64][520];    // 64 x 512 (+8 pad)  ~66.5 KB
  __shared__ __align__(16) float Ssa[64][68];    // scores, k-half 0   ~17.4 KB
  __shared__ __align__(16) float Ssb[64][68];    // scores, k-half 1   ~17.4 KB
  __shared__ __align__(16) bf16  Ps[64][72];     // probs 64x64 (+8)   ~9.2 KB
  __shared__ float mrow[64], lrow[64], srow[64];

  const int t     = threadIdx.x;
  const int lane  = t & 31;
  const int w     = t >> 5;
  const int h     = blockIdx.y;
  const int rbase = blockIdx.x * 64;
  const size_t hbase = (size_t)h << 20;          // h * N * D

  // async-stage the 64x512 Q block into LDS (GLOBAL_LOAD_ASYNC_TO_LDS_B128)
#pragma unroll
  for (int it = 0; it < 16; ++it) {
    int idx = (it * 256 + t) * 8;
    int r = idx >> 9;
    int c = idx & 511;
    async_copy_b128(&Qs[r][c], Q + hbase + (size_t)(rbase + r) * 512 + c);
  }
  if (t < 64) { mrow[t] = -3.0e38f; lrow[t] = 0.0f; srow[t] = 0.0f; }
  wait_async0();
  __syncthreads();

  const int mt  = (w & 3) * 16;        // stage-1: m-tile base
  const int kh  = (w >> 2) * 8;        // stage-1: k-half (8 k-steps of 32)
  const int dt0 = w * 64;              // stage-2: unique d-slice base
  const int cn  = lane & 15;
  const int hi  = (lane >> 4) & 1;

  v8f oacc[4][4] = {};                 // [row-tile][d-tile] 16x16 f32 tiles

  for (int m0 = 0; m0 < NN; m0 += 64) {
    // prefetch this chunk's bias block (64 rows x 64 cols, f32)
    __builtin_prefetch(biasM + (size_t)(rbase + (t >> 2)) * 2048 + m0 + (t & 3) * 16, 0, 1);

    // ---- stage 1: partial scores, dup=1 on K fragments ----
    v8f sacc[4] = {};
#pragma unroll
    for (int kk = 0; kk < 8; ++kk) {
      int kks = (kh + kk) * 32;
      v16bf b = load_b_frag(Km + hbase + (size_t)(m0 + mt) * 512 + kks, 512, lane);
#pragma unroll
      for (int rt = 0; rt < 4; ++rt) {
        v16bf a = load_a_frag(&Qs[rt * 16][kks], 520, lane);
        sacc[rt] = WMMA_BF16(a, b, sacc[rt]);
      }
    }
    {
      float (*Sdst)[68] = (w >> 2) ? Ssb : Ssa;
#pragma unroll
      for (int rt = 0; rt < 4; ++rt)
#pragma unroll
        for (int e = 0; e < 8; ++e)
          Sdst[rt * 16 + e + 8 * hi][mt + cn] = sacc[rt][e];
    }
    __syncthreads();

    // ---- online softmax: 4 threads per row, 16 elements each ----
    {
      int row = t >> 2;
      int seg = (t & 3) * 16;
      const float* brow = biasM + (size_t)(rbase + row) * 2048 + m0 + seg;
      float vals[16], vmax = -3.0e38f;
#pragma unroll
      for (int j = 0; j < 16; ++j) {
        vals[j] = Ssa[row][seg + j] + Ssb[row][seg + j] + brow[j];
        vmax = fmaxf(vmax, vals[j]);
      }
      vmax = fmaxf(vmax, __shfl_xor(vmax, 1));
      vmax = fmaxf(vmax, __shfl_xor(vmax, 2));
      float mold = mrow[row];
      float mnew = fmaxf(mold, vmax);
      float lsum = 0.0f;
#pragma unroll
      for (int j = 0; j < 16; ++j) {
        float pv = __expf(vals[j] - mnew);
        lsum += pv;
        ((unsigned short*)&Ps[row][0])[seg + j] = f2bf_bits(pv);
      }
      lsum += __shfl_xor(lsum, 1);
      lsum += __shfl_xor(lsum, 2);
      if ((t & 3) == 0) {
        float fac = __expf(mold - mnew);
        srow[row] = fac;
        lrow[row] = lrow[row] * fac + lsum;
        mrow[row] = mnew;
      }
    }
    __syncthreads();

    // ---- stage 2: rescale O, then O += P @ Vchunk (dup=1 on Vt fragments) ----
#pragma unroll
    for (int rt = 0; rt < 4; ++rt)
#pragma unroll
      for (int e = 0; e < 8; ++e) {
        float fac = srow[rt * 16 + e + 8 * hi];
#pragma unroll
        for (int dj = 0; dj < 4; ++dj) oacc[rt][dj][e] *= fac;
      }
#pragma unroll
    for (int km = 0; km < 2; ++km) {
      v16bf a[4];
#pragma unroll
      for (int rt = 0; rt < 4; ++rt)
        a[rt] = load_a_frag(&Ps[rt * 16][km * 32], 72, lane);
#pragma unroll
      for (int dj = 0; dj < 4; ++dj) {
        v16bf b = load_b_frag(Vt + hbase + (size_t)(dt0 + dj * 16) * 2048 + m0 + km * 32,
                              2048, lane);
#pragma unroll
        for (int rt = 0; rt < 4; ++rt)
          oacc[rt][dj] = WMMA_BF16(a[rt], b, oacc[rt][dj]);
      }
    }
    // no barrier needed here: next stage-1 only writes Ssa/Ssb (not read now),
    // and Ps/srow are rewritten only after the post-stage-1 barrier, by which
    // time every wave has finished this stage 2.
  }

  // ---- finalize: divide by l, write AO[n][h*512+d] ----
#pragma unroll
  for (int rt = 0; rt < 4; ++rt)
#pragma unroll
    for (int e = 0; e < 8; ++e) {
      int r = rt * 16 + e + 8 * hi;
      float inv = 1.0f / lrow[r];
#pragma unroll
      for (int dj = 0; dj < 4; ++dj) {
        int c = dt0 + dj * 16 + cn;
        ((unsigned short*)AO)[(size_t)(rbase + r) * 4096 + h * 512 + c] =
            f2bf_bits(oacc[rt][dj][e] * inv);
      }
    }
}

// ---------------------------------------------------------------- launch
extern "C" void kernel_launch(void* const* d_in, const int* in_sizes, int n_in,
                              void* d_out, int out_size, void* d_ws, size_t ws_size,
                              hipStream_t stream) {
  const float* x  = (const float*)d_in[0];
  const float* sp = (const float*)d_in[1];
  const float* ed = (const float*)d_in[2];
  const float* Wq = (const float*)d_in[3];
  const float* Wk = (const float*)d_in[4];
  const float* Wv = (const float*)d_in[5];
  const float* Wo = (const float*)d_in[6];

  char* p = (char*)d_ws;
  bf16*  x16  = (bf16*)p;  p += (size_t)2048 * 512 * 2;
  bf16*  Wqt  = (bf16*)p;  p += (size_t)4096 * 512 * 2;
  bf16*  Wkt  = (bf16*)p;  p += (size_t)4096 * 512 * 2;
  bf16*  Wvt  = (bf16*)p;  p += (size_t)4096 * 512 * 2;
  bf16*  Wot  = (bf16*)p;  p += (size_t)512 * 4096 * 2;
  float* biasM = (float*)p; p += (size_t)2048 * 2048 * 4;
  bf16*  Q16  = (bf16*)p;  p += (size_t)2048 * 4096 * 2;
  bf16*  K16  = (bf16*)p;  p += (size_t)2048 * 4096 * 2;
  bf16*  Vt16 = (bf16*)p;  p += (size_t)2048 * 4096 * 2;
  bf16*  AO16 = (bf16*)p;  p += (size_t)2048 * 4096 * 2;

  cvt_kernel<<<1024, 256, 0, stream>>>(x, x16, 2048 * 512);
  cvt_tr_kernel<<<2048, 256, 0, stream>>>(Wq, Wqt, 512, 4096);
  cvt_tr_kernel<<<2048, 256, 0, stream>>>(Wk, Wkt, 512, 4096);
  cvt_tr_kernel<<<2048, 256, 0, stream>>>(Wv, Wvt, 512, 4096);
  cvt_tr_kernel<<<2048, 256, 0, stream>>>(Wo, Wot, 4096, 512);
  bias_kernel<<<4096, 256, 0, stream>>>(sp, ed, biasM, 2048 * 2048);

  const float scale = 0.044194173824159216f;  // 1/sqrt(512), folded into Q
  dim3 blk(256);
  dim3 gqkv(4096 / 256, 2048 / 64);
  gemm_bf16_kernel<0><<<gqkv, blk, 0, stream>>>(x16, Wqt, Q16, 2048, 4096, 512, scale);
  gemm_bf16_kernel<0><<<gqkv, blk, 0, stream>>>(x16, Wkt, K16, 2048, 4096, 512, 1.0f);
  gemm_bf16_kernel<2><<<gqkv, blk, 0, stream>>>(x16, Wvt, Vt16, 2048, 4096, 512, 1.0f);

  attn_kernel<<<dim3(32, 8), blk, 0, stream>>>(Q16, K16, Vt16, biasM, AO16);

  dim3 gout(512 / 256, 2048 / 64);
  gemm_bf16_kernel<1><<<gout, blk, 0, stream>>>(AO16, Wot, d_out, 2048, 512, 4096, 1.0f);
}
